// WaveletLoss_28372553957611
// MI455X (gfx1250) — compile-verified, matrix operand order checked by reading, not run on hardware
//
#include <hip/hip_runtime.h>

typedef __attribute__((ext_vector_type(16))) _Float16 v16h;
typedef __attribute__((ext_vector_type(8)))  float    v8f;

// Layout constants for [8,3,1024,1024] f32: 24 contiguous 1024x1024 images.
// One "work item" = one float4 (4 cols) of an even row + matching odd row
// = two 2x2 Haar blocks = 8 |butterfly| values.
#define WL_W        1024      // image width (floats)
#define WL_F4_ROW   256       // float4s per row
#define WL_PITCH2   2048      // floats per row-pair

__global__ __launch_bounds__(256)
void wl_partial(const float* __restrict__ sr, const float* __restrict__ gt,
                float* __restrict__ partial, int M4)
{
    const int tid = blockIdx.x * blockDim.x + threadIdx.x;
    const int T   = gridDim.x * blockDim.x;
    // Uniform trip count across the whole grid -> EXEC all-1s at the WMMA.
    const int niter = (M4 + 2 * T - 1) / (2 * T);

    v16h ones;
#pragma unroll
    for (int i = 0; i < 16; ++i) ones[i] = (_Float16)1.0f;

    v8f c = {};   // f32 accumulator fragment

    for (int k = 0; k < niter; ++k) {
        float u[16];
#pragma unroll
        for (int h = 0; h < 2; ++h) {
            const int w = tid + (2 * k + h) * T;     // work-item id
            float* uu = u + 8 * h;
            if (w < M4) {
                const int rp = w >> 8;               // row-pair index
                const int c4 = w & (WL_F4_ROW - 1);  // float4 column
                const int eb = rp * WL_PITCH2 + c4 * 4;
                const float4 es = *(const float4*)(sr + eb);
                const float4 os = *(const float4*)(sr + eb + WL_W);
                const float4 eg = *(const float4*)(gt + eb);
                const float4 og = *(const float4*)(gt + eb + WL_W);
                // block 0: a=x[2i][2j], b=x[2i][2j+1], c=x[2i+1][2j], d=x[2i+1][2j+1]
                float da = es.x - eg.x, db = es.y - eg.y;
                float dc = os.x - og.x, dd = os.y - og.y;
                float p = da + db, q = dc + dd, r = da - db, t = dc - dd;
                uu[0] = fabsf(p + q);   // 2*|cA diff|
                uu[1] = fabsf(p - q);   // 2*|cH diff|
                uu[2] = fabsf(r + t);   // 2*|cV diff|
                uu[3] = fabsf(r - t);   // 2*|cD diff|
                // block 1
                da = es.z - eg.z; db = es.w - eg.w;
                dc = os.z - og.z; dd = os.w - og.w;
                p = da + db; q = dc + dd; r = da - db; t = dc - dd;
                uu[4] = fabsf(p + q);
                uu[5] = fabsf(p - q);
                uu[6] = fabsf(r + t);
                uu[7] = fabsf(r - t);
            } else {
#pragma unroll
                for (int i = 0; i < 8; ++i) uu[i] = 0.0f;
            }
        }
        // D[m][n] = C[m][n] + sum_k A[m][k] with B == ones:
        // one WMMA folds 512 lane values into the f32 accumulator.
        v16h a;
#pragma unroll
        for (int i = 0; i < 16; ++i) a[i] = (_Float16)u[i];
        c = __builtin_amdgcn_wmma_f32_16x16x32_f16(
                /*neg_a=*/false, a, /*neg_b=*/false, ones,
                /*c_mod=*/(short)0, c, /*reuse_a=*/false, /*reuse_b=*/false);
    }

    // Lane n (0..15) holds rows 0..7 of column n; lanes 16..31 hold rows 8..15.
    // Columns are replicated, so: half-sum per lane, then add across wave halves.
    float s = c[0] + c[1] + c[2] + c[3] + c[4] + c[5] + c[6] + c[7];
    s += __shfl_xor(s, 16, 32);

    __shared__ float lds[8];
    const int lane = threadIdx.x & 31;
    const int wv   = threadIdx.x >> 5;
    if (lane == 0) lds[wv] = s;
    __syncthreads();
    if (threadIdx.x == 0) {
        float t = 0.0f;
#pragma unroll
        for (int i = 0; i < 8; ++i) t += lds[i];   // fixed order -> deterministic
        partial[blockIdx.x] = t;
    }
}

__global__ __launch_bounds__(256)
void wl_finish(const float* __restrict__ partial, int G, float scale,
               float* __restrict__ out)
{
    __shared__ float lds[256];
    float acc = 0.0f;
    for (int i = threadIdx.x; i < G; i += 256) acc += partial[i];  // fixed order
    lds[threadIdx.x] = acc;
    __syncthreads();
    for (int s = 128; s > 0; s >>= 1) {
        if (threadIdx.x < (unsigned)s) lds[threadIdx.x] += lds[threadIdx.x + s];
        __syncthreads();
    }
    if (threadIdx.x == 0) out[0] = lds[0] * scale;
}

extern "C" void kernel_launch(void* const* d_in, const int* in_sizes, int n_in,
                              void* d_out, int out_size, void* d_ws, size_t ws_size,
                              hipStream_t stream)
{
    const float* sr = (const float*)d_in[0];
    const float* gt = (const float*)d_in[1];
    float* out      = (float*)d_out;
    float* partial  = (float*)d_ws;

    const int N  = in_sizes[0];   // 8*3*1024*1024 = 25,165,824 floats per tensor
    const int M4 = N / 8;         // work items (2 Haar blocks each) = 3,145,728

    int grid = 6144;              // 1.57M threads; 2T == M4 -> exactly 1 WMMA body/thread
    if ((size_t)grid * sizeof(float) > ws_size) {
        grid = (int)(ws_size / sizeof(float));
        if (grid < 1) grid = 1;
    }

    wl_partial<<<grid, 256, 0, stream>>>(sr, gt, partial, M4);

    // loss = (0.5 * sum_all_u) / (N/4) = sum_all_u * (2/N)
    const float scale = 2.0f / (float)N;
    wl_finish<<<1, 256, 0, stream>>>(partial, grid, scale, out);
}
